// NSTLoss_65850438582860
// MI455X (gfx1250) — matrix operand order repaired and checked by reading, not compiled, not run on hardware
//
#include <hip/hip_runtime.h>

typedef __attribute__((ext_vector_type(16))) __bf16 bf16x16;
typedef __attribute__((ext_vector_type(8)))  float  floatx8;

#define B_DIM   8
#define C_DIM   1024
#define HW_DIM  3136
#define KSTEP   64
#define LDS_STRIDE 144            // 64 bf16 = 128B per tile row + 16B pad (conflict-free b128)
#define TILE_BYTES (128 * LDS_STRIDE)   // 18432 B per tile
#define BUF_BYTES  (2 * TILE_BYTES)     // A+B tiles per buffer

__device__ inline unsigned short f2bf(float f) {
  unsigned int u = __float_as_uint(f);
  u += 0x7FFFu + ((u >> 16) & 1u);      // round-to-nearest-even
  return (unsigned short)(u >> 16);
}

__global__ void nst_init(float* accs) {
  if (threadIdx.x < 2) accs[threadIdx.x] = 0.f;
}

// One wave per row: compute 1/max(||row||,eps), write bf16 normalized row.
__global__ __launch_bounds__(256) void nst_normalize(const float* __restrict__ fm_s,
                                                     const float* __restrict__ fm_t,
                                                     unsigned short* __restrict__ sBF,
                                                     unsigned short* __restrict__ tBF) {
  const int wave = threadIdx.x >> 5, lane = threadIdx.x & 31;
  const int row = blockIdx.x * 8 + wave;              // 0 .. 2*B*C-1
  const float* src;
  unsigned short* dst;
  if (row < B_DIM * C_DIM) {
    src = fm_s + (size_t)row * HW_DIM;
    dst = sBF  + (size_t)row * HW_DIM;
  } else {
    const int r = row - B_DIM * C_DIM;
    src = fm_t + (size_t)r * HW_DIM;
    dst = tBF  + (size_t)r * HW_DIM;
  }
  const float4* s4 = (const float4*)src;
  float acc = 0.f;
  for (int i = lane; i < HW_DIM / 4; i += 32) {
    float4 v = s4[i];
    acc = fmaf(v.x, v.x, acc); acc = fmaf(v.y, v.y, acc);
    acc = fmaf(v.z, v.z, acc); acc = fmaf(v.w, v.w, acc);
  }
  #pragma unroll
  for (int off = 16; off; off >>= 1) acc += __shfl_xor(acc, off, 32);
  const float scale = 1.f / fmaxf(sqrtf(acc), 1e-12f);
  ushort4* d4 = (ushort4*)dst;
  for (int i = lane; i < HW_DIM / 4; i += 32) {
    float4 v = s4[i];
    ushort4 o;
    o.x = f2bf(v.x * scale); o.y = f2bf(v.y * scale);
    o.z = f2bf(v.z * scale); o.w = f2bf(v.w * scale);
    d4[i] = o;
  }
}

// Fused tile-GEMM + sum-of-squares reduction, double-buffered LDS pipeline.
// blockIdx.x: 64 output tiles (8x8 of 128x128), .y: batch, .z: 0 => S*S^T, 1 => T*S^T
__global__ __launch_bounds__(256) void nst_gemm(const unsigned short* __restrict__ sBF,
                                                const unsigned short* __restrict__ tBF,
                                                float* __restrict__ accs) {
  __shared__ uint4 ldsBuf[2 * BUF_BYTES / 16];    // 73728 B (double-buffered A+B tiles)
  __shared__ float red[8];
  unsigned char* lds = (unsigned char*)ldsBuf;

  const int tid   = threadIdx.x;
  const int z     = blockIdx.z;
  const int b     = blockIdx.y;
  const int mBase = (int)(blockIdx.x >> 3) * 128;
  const int nBase = (int)(blockIdx.x & 7) * 128;
  const unsigned short* X = z ? tBF : sBF;   // A-operand rows
  const unsigned short* S = sBF;             // B-operand rows (columns of Gram)

  // Cooperative global->LDS mapping: 2 threads per tile row, 64B each.
  const int gRow  = tid >> 1;
  const int gHalf = (tid & 1) * 32;          // bf16 elements
  const unsigned short* gA = X + ((size_t)b * C_DIM + mBase + gRow) * HW_DIM + gHalf;
  const unsigned short* gB = S + ((size_t)b * C_DIM + nBase + gRow) * HW_DIM + gHalf;
  const int stOffA = gRow * LDS_STRIDE + gHalf * 2;              // within A tile
  const int stOffB = TILE_BYTES + gRow * LDS_STRIDE + gHalf * 2; // within buffer

  // Wave tiling: 4 waves in M x 2 waves in N; each wave: 2 M-tiles x 4 N-tiles.
  const int wave = tid >> 5, lane = tid & 31;
  const int waveM = wave & 3, waveN = wave >> 2;
  const int laneLo = lane & 15;
  const int hi = lane >> 4;

  int aOff[2], bOff[4];
  #pragma unroll
  for (int mt = 0; mt < 2; ++mt)
    aOff[mt] = (waveM * 32 + mt * 16 + laneLo) * LDS_STRIDE + hi * 32;
  #pragma unroll
  for (int nt = 0; nt < 4; ++nt)
    bOff[nt] = TILE_BYTES + (waveN * 64 + nt * 16 + laneLo) * LDS_STRIDE + hi * 32;

  floatx8 acc[2][4];
  const floatx8 z8 = {0.f, 0.f, 0.f, 0.f, 0.f, 0.f, 0.f, 0.f};
  #pragma unroll
  for (int mt = 0; mt < 2; ++mt)
    #pragma unroll
    for (int nt = 0; nt < 4; ++nt) acc[mt][nt] = z8;

  union Frag { bf16x16 v; uint4 q[2]; };

  // Compute 16 WMMAs on one LDS buffer; all fragment loads batched before use.
  auto compute = [&](const unsigned char* buf) {
    #pragma unroll
    for (int kk = 0; kk < 2; ++kk) {
      const int kByte = kk * 64;
      Frag a[2], bb[4];
      #pragma unroll
      for (int mt = 0; mt < 2; ++mt) {
        a[mt].q[0] = *(const uint4*)(buf + aOff[mt] + kByte);
        a[mt].q[1] = *(const uint4*)(buf + aOff[mt] + kByte + 16);
      }
      #pragma unroll
      for (int nt = 0; nt < 4; ++nt) {
        bb[nt].q[0] = *(const uint4*)(buf + bOff[nt] + kByte);
        bb[nt].q[1] = *(const uint4*)(buf + bOff[nt] + kByte + 16);
      }
      #pragma unroll
      for (int nt = 0; nt < 4; ++nt)
        #pragma unroll
        for (int mt = 0; mt < 2; ++mt)
          acc[mt][nt] = __builtin_amdgcn_wmma_f32_16x16x32_bf16(
              false, a[mt].v, false, bb[nt].v, (short)0, acc[mt][nt], false, false);
    }
  };

  // Prologue: stage tile 0 into buffer 0.
  {
    uint4 pa0 = *(const uint4*)(gA + 0),  pa1 = *(const uint4*)(gA + 8);
    uint4 pa2 = *(const uint4*)(gA + 16), pa3 = *(const uint4*)(gA + 24);
    uint4 pb0 = *(const uint4*)(gB + 0),  pb1 = *(const uint4*)(gB + 8);
    uint4 pb2 = *(const uint4*)(gB + 16), pb3 = *(const uint4*)(gB + 24);
    uint4* lA = (uint4*)(lds + stOffA);
    uint4* lB = (uint4*)(lds + stOffB);
    lA[0] = pa0; lA[1] = pa1; lA[2] = pa2; lA[3] = pa3;
    lB[0] = pb0; lB[1] = pb1; lB[2] = pb2; lB[3] = pb3;
  }
  __syncthreads();

  const int nSteps = HW_DIM / KSTEP;   // 49
  for (int i = 0; i < nSteps - 1; ++i) {
    // Issue next tile's global loads first (latency hidden by WMMAs below).
    gA += KSTEP; gB += KSTEP;
    uint4 pa0 = *(const uint4*)(gA + 0),  pa1 = *(const uint4*)(gA + 8);
    uint4 pa2 = *(const uint4*)(gA + 16), pa3 = *(const uint4*)(gA + 24);
    uint4 pb0 = *(const uint4*)(gB + 0),  pb1 = *(const uint4*)(gB + 8);
    uint4 pb2 = *(const uint4*)(gB + 16), pb3 = *(const uint4*)(gB + 24);

    compute(lds + (i & 1) * BUF_BYTES);

    unsigned char* nbuf = lds + ((i + 1) & 1) * BUF_BYTES;
    uint4* lA = (uint4*)(nbuf + stOffA);
    uint4* lB = (uint4*)(nbuf + stOffB);
    lA[0] = pa0; lA[1] = pa1; lA[2] = pa2; lA[3] = pa3;
    lB[0] = pb0; lB[1] = pb1; lB[2] = pb2; lB[3] = pb3;
    __syncthreads();
  }
  compute(lds + ((nSteps - 1) & 1) * BUF_BYTES);

  // Local sum of squares of this block's 128x128 tile, then one atomic.
  float local = 0.f;
  #pragma unroll
  for (int mt = 0; mt < 2; ++mt)
    #pragma unroll
    for (int nt = 0; nt < 4; ++nt)
      #pragma unroll
      for (int j = 0; j < 8; ++j) {
        float e = acc[mt][nt][j];
        local = fmaf(e, e, local);
      }
  #pragma unroll
  for (int off = 16; off; off >>= 1) local += __shfl_xor(local, off, 32);
  if (lane == 0) red[wave] = local;
  __syncthreads();
  if (tid == 0) {
    float t = 0.f;
    #pragma unroll
    for (int i = 0; i < 8; ++i) t += red[i];
    atomicAdd(&accs[z], t);
  }
}

__global__ void nst_finalize(const float* __restrict__ accs, float* __restrict__ out) {
  out[0] = (accs[0] - 2.f * accs[1]) * (1.f / 8388608.f);   // / (B*C*C)
}

extern "C" void kernel_launch(void* const* d_in, const int* in_sizes, int n_in,
                              void* d_out, int out_size, void* d_ws, size_t ws_size,
                              hipStream_t stream) {
  const float* fm_s = (const float*)d_in[0];
  const float* fm_t = (const float*)d_in[1];
  float* out = (float*)d_out;

  unsigned char* ws = (unsigned char*)d_ws;
  float* accs = (float*)ws;                                   // 2 floats (+pad)
  unsigned short* sBF = (unsigned short*)(ws + 256);
  unsigned short* tBF = sBF + (size_t)B_DIM * C_DIM * HW_DIM;

  nst_init<<<1, 32, 0, stream>>>(accs);
  nst_normalize<<<(2 * B_DIM * C_DIM) / 8, 256, 0, stream>>>(fm_s, fm_t, sBF, tBF);
  dim3 grid(64, B_DIM, 2);
  nst_gemm<<<grid, 256, 0, stream>>>(sBF, tBF, accs);
  nst_finalize<<<1, 1, 0, stream>>>(accs, out);
}
